// AliBiFlashAttention_56573309223557
// MI455X (gfx1250) — matrix operand order repaired and checked by reading, not compiled, not run on hardware
//
#include <hip/hip_runtime.h>
#include <hip/hip_bf16.h>

#define EMBED   1024
#define NHEADS  16
#define HDIM    64
#define BATCH   2
#define SEQ     2048
#define BSROWS  (BATCH*SEQ)   // 4096

typedef __bf16 bf16_t;
typedef __attribute__((ext_vector_type(16))) __bf16 v16bf;
typedef __attribute__((ext_vector_type(8)))  __bf16 v8bf;
typedef __attribute__((ext_vector_type(8)))  float  v8f;

// ---------------------------------------------------------------- helpers

__device__ __forceinline__ v8f wmma_bf16(v16bf a, v16bf b, v8f c) {
  return __builtin_amdgcn_wmma_f32_16x16x32_bf16(
      /*neg_a=*/false, a, /*neg_b=*/false, b,
      /*c_mod=*/(short)0, c, /*reuse_a=*/false, /*reuse_b=*/false);
}

// Async global->LDS copy, 16 bytes per lane. Tracked by ASYNCcnt.
__device__ __forceinline__ void async_b128(void* lds_ptr, const void* gptr) {
  unsigned lo = (unsigned)(uintptr_t)lds_ptr;          // LDS byte address (low 32b)
  unsigned long long ga = (unsigned long long)(uintptr_t)gptr;
  asm volatile("global_load_async_to_lds_b128 %0, %1, off"
               :: "v"(lo), "v"(ga) : "memory");
}
__device__ __forceinline__ void wait_async0() {
  asm volatile("s_wait_asynccnt 0x0" ::: "memory");
}

__device__ __forceinline__ float redmax16(float v) {
#pragma unroll
  for (int m = 1; m < 16; m <<= 1) v = fmaxf(v, __shfl_xor(v, m, 32));
  return v;
}
__device__ __forceinline__ float redsum16(float v) {
#pragma unroll
  for (int m = 1; m < 16; m <<= 1) v += __shfl_xor(v, m, 32);
  return v;
}

// ---------------------------------------------------------------- converts

__global__ void cvt_f32_to_bf16(const float* __restrict__ src,
                                bf16_t* __restrict__ dst, int n) {
  int i = blockIdx.x * blockDim.x + threadIdx.x;
  if (i < n) dst[i] = (bf16_t)src[i];
}

// Pack W (for y = x @ W^T) into B-fragment friendly panels:
//   Bp[(k0/32)*(E*32) + n*32 + kk] = W[n*E + (k0 + kk)]
// A GEMM B-tile (k0, n0..n0+64) is then 4KB *contiguous* -> pure async copy,
// and the WMMA B-fragment read from LDS is contiguous too.
__global__ void cvt_pack_weight(const float* __restrict__ W,
                                bf16_t* __restrict__ Bp, int E) {
  int i = blockIdx.x * blockDim.x + threadIdx.x;
  if (i < E * E) {
    int kt  = i / (E * 32);
    int rem = i - kt * E * 32;
    int n   = rem >> 5;
    int kk  = rem & 31;
    Bp[i] = (bf16_t)W[n * E + kt * 32 + kk];
  }
}

// ---------------------------------------------------------------- GEMM
// C[M,N] = A[M,K](bf16 row-major) x Bp(packed panels) + bias[N]
// block tile 64x64, 8 waves; wave (wr,wc) owns rows wr*16..+16, cols wc*32..+32.
// Double-buffered LDS, staged entirely with async global->LDS b128 copies.
// MODE: 0 = f32 row-major out, 1 = bf16 row-major out,
//       2 = bf16 out transposed per head: VTg[((b*H+h)*D+d)*S + s]

template <int MODE>
__global__ __launch_bounds__(256)
void gemm_bf16_wmma(const bf16_t* __restrict__ A, const bf16_t* __restrict__ Bp,
                    const float* __restrict__ bias,
                    float* __restrict__ Cf, bf16_t* __restrict__ Cb,
                    int M, int N, int K) {
  __shared__ bf16_t ldsA[2][64][32];     // [buf][m][k]
  __shared__ bf16_t ldsB[2][64 * 32];    // [buf][n*32 + k] (packed panel image)

  const int t    = threadIdx.x;
  const int wave = t >> 5;
  const int lane = t & 31;
  const int lh   = lane >> 4;
  const int ll   = lane & 15;
  const int wr   = wave >> 1;
  const int wc   = wave & 1;
  const int m0   = blockIdx.y * 64;
  const int n0   = blockIdx.x * 64;

  const int ra = t >> 2;                 // A stage row      (0..63)
  const int ca = (t & 3) * 8;            // A stage k-chunk  (0,8,16,24)

  v8f c0 = {}; v8f c1 = {};

  // ---- software pipeline: stage(k0 -> buf)
  auto stage = [&](int k0, int buf) {
    async_b128(&ldsA[buf][ra][ca], A + (long)(m0 + ra) * K + k0 + ca);
    async_b128(&ldsB[buf][t * 8],
               Bp + (long)(k0 >> 5) * N * 32 + (long)n0 * 32 + t * 8);
  };

  stage(0, 0);
  wait_async0();
  __syncthreads();

  int buf = 0;
  for (int k0 = 0; k0 < K; k0 += 32) {
    if (k0 + 32 < K) stage(k0 + 32, buf ^ 1);

    // A fragment: lane = row; elems 0..15 -> k=lh*16.. (contiguous)
    const int ar = wr * 16 + ll;
    v16bf af = *(const v16bf*)(&ldsA[buf][ar][lh * 16]);

    // B fragments: lane = col, elems = k rows (lh*16..+16), contiguous
    v16bf bf0 = *(const v16bf*)(&ldsB[buf][(wc * 32 + ll) * 32 + lh * 16]);
    v16bf bf1 = *(const v16bf*)(&ldsB[buf][(wc * 32 + 16 + ll) * 32 + lh * 16]);

    c0 = wmma_bf16(af, bf0, c0);
    c1 = wmma_bf16(af, bf1, c1);

    wait_async0();
    __syncthreads();
    buf ^= 1;
  }

  // Epilogue: elem i -> row = wr*16 + lh*8 + i, col = ll (+16 for c1)
  const int col0 = n0 + wc * 32 + ll;
  const int col1 = col0 + 16;
  const float b0 = bias[col0];
  const float b1 = bias[col1];
  const int rbase = m0 + wr * 16 + lh * 8;
#pragma unroll
  for (int i = 0; i < 8; ++i) {
    const long r = rbase + i;
    float v0 = c0[i] + b0;
    float v1 = c1[i] + b1;
    if (MODE == 0) {
      Cf[r * N + col0] = v0;
      Cf[r * N + col1] = v1;
    } else if (MODE == 1) {
      Cb[r * N + col0] = (bf16_t)v0;
      Cb[r * N + col1] = (bf16_t)v1;
    } else {
      // V output, transposed per head for async attention staging
      const long bb = r >> 11;           // r / SEQ
      const long s  = r & (SEQ - 1);
      const long h0 = col0 >> 6, d0 = col0 & 63;
      const long h1 = col1 >> 6, d1 = col1 & 63;
      Cb[((bb * NHEADS + h0) * HDIM + d0) * SEQ + s] = (bf16_t)v0;
      Cb[((bb * NHEADS + h1) * HDIM + d1) * SEQ + s] = (bf16_t)v1;
    }
  }
}

// ---------------------------------------------------------------- attention
// One block = (b, h, 128 q rows); 8 waves, each wave owns 16 q rows.
// Flash loop over 32-key tiles; V tiles async-staged from transposed VTg.

__global__ __launch_bounds__(256)
void alibi_flash_attn(const bf16_t* __restrict__ Q, const bf16_t* __restrict__ Kmat,
                      const bf16_t* __restrict__ VTg, bf16_t* __restrict__ O) {
  __shared__ bf16_t VT[64][32];        // V tile transposed: [d][key]
  __shared__ bf16_t P[8][16][32];      // per-wave probs tile [qrow][key]

  const int t    = threadIdx.x;
  const int wave = t >> 5;
  const int lane = t & 31;
  const int lh   = lane >> 4;
  const int ll   = lane & 15;

  const int blk = blockIdx.x;
  const int qt  = blk & 15;
  const int h   = (blk >> 4) & 15;
  const int b   = blk >> 8;

  const int q0b  = qt * 128;
  const int q0   = q0b + wave * 16;
  const int qmax = q0 + 15;

  const float slope = exp2f(-0.5f * (float)(h + 1));  // ALiBi, H=16 (pow2)
  const float inv_sqrt_d = 0.125f;                    // 1/sqrt(64)

  const long headoff = (long)h * HDIM;
  const long batoff  = (long)b * SEQ * EMBED;
  const bf16_t* Vhead = VTg + ((long)b * NHEADS + h) * HDIM * SEQ;  // [d][s]

  const int vd = t >> 2;               // V stage d-row   (0..63)
  const int vk = (t & 3) * 8;          // V stage key-chunk

  // Q A-fragments for d=0..31 and d=32..63 (loaded once)
  v16bf qa0, qa1;
  {
    const long ro = batoff + (long)(q0 + ll) * EMBED + headoff;
    qa0 = *(const v16bf*)(Q + ro + lh * 16);
    qa1 = *(const v16bf*)(Q + ro + 32 + lh * 16);
  }

  float mrow[8], lrow[8];
  v8f acc[4] = {v8f{}, v8f{}, v8f{}, v8f{}};
#pragma unroll
  for (int i = 0; i < 8; ++i) { mrow[i] = -1e30f; lrow[i] = 0.f; }

  const int nkt = (q0b + 128) / 32;    // causal upper bound for whole block
  for (int kt = 0; kt < nkt; ++kt) {
    const int k0 = kt * 32;
    __syncthreads();                   // prior reads of VT finished
    async_b128(&VT[vd][vk], Vhead + (long)vd * SEQ + k0 + vk);
    wait_async0();
    __syncthreads();                   // VT visible to all waves

    if (k0 <= qmax) {
      // ---- scores: two 16x16 tiles (keys k0..+16, k0+16..+32)
      v8f s0 = {}, s1 = {};
      {
        const long kr0 = batoff + (long)(k0 + ll) * EMBED + headoff;
        v16bf kb00 = *(const v16bf*)(Kmat + kr0 + lh * 16);       // d 0..31
        v16bf kb01 = *(const v16bf*)(Kmat + kr0 + 32 + lh * 16);  // d 32..63
        s0 = wmma_bf16(qa0, kb00, s0);
        s0 = wmma_bf16(qa1, kb01, s0);
        const long kr1 = kr0 + (long)16 * EMBED;
        v16bf kb10 = *(const v16bf*)(Kmat + kr1 + lh * 16);
        v16bf kb11 = *(const v16bf*)(Kmat + kr1 + 32 + lh * 16);
        s1 = wmma_bf16(qa0, kb10, s1);
        s1 = wmma_bf16(qa1, kb11, s1);
      }

      // ---- ALiBi + causal + online softmax
      const int jc0 = k0 + ll;
      const int jc1 = k0 + 16 + ll;
#pragma unroll
      for (int i = 0; i < 8; ++i) {
        const int qrow = q0 + lh * 8 + i;
        float v0 = s0[i] * inv_sqrt_d - slope * (float)(qrow - jc0);
        if (jc0 > qrow) v0 = -1e30f;
        float v1 = s1[i] * inv_sqrt_d - slope * (float)(qrow - jc1);
        if (jc1 > qrow) v1 = -1e30f;

        float mx = redmax16(fmaxf(v0, v1));
        float mnew = fmaxf(mrow[i], mx);
        float sc = __expf(mrow[i] - mnew);
        float p0 = __expf(v0 - mnew);
        float p1 = __expf(v1 - mnew);
        lrow[i] = lrow[i] * sc + redsum16(p0 + p1);
        mrow[i] = mnew;
        acc[0][i] *= sc; acc[1][i] *= sc; acc[2][i] *= sc; acc[3][i] *= sc;

        P[wave][lh * 8 + i][ll]      = (bf16_t)p0;
        P[wave][lh * 8 + i][16 + ll] = (bf16_t)p1;
      }
      // wave-local LDS RAW: DS ops are in-order per wave; block compiler reorder
      asm volatile("s_wait_dscnt 0x0" ::: "memory");

      // probs A-fragment (contraction over 32 keys)
      v16bf pf = *(const v16bf*)(&P[wave][ll][lh * 16]);

      // PV: 4 output chunks of 16 columns (d)
#pragma unroll
      for (int c2 = 0; c2 < 4; ++c2) {
        v16bf vf = *(const v16bf*)(&VT[c2 * 16 + ll][lh * 16]);
        acc[c2] = wmma_bf16(pf, vf, acc[c2]);
      }
    }
  }

  // ---- epilogue: normalize and store bf16 (row-major, head-interleaved)
#pragma unroll
  for (int i = 0; i < 8; ++i) {
    const float inv_l = 1.0f / lrow[i];
    const long row = batoff + (long)(q0 + lh * 8 + i) * EMBED + headoff;
#pragma unroll
    for (int c2 = 0; c2 < 4; ++c2)
      O[row + c2 * 16 + ll] = (bf16_t)(acc[c2][i] * inv_l);
  }
}

// ---------------------------------------------------------------- launch

extern "C" void kernel_launch(void* const* d_in, const int* in_sizes, int n_in,
                              void* d_out, int out_size, void* d_ws, size_t ws_size,
                              hipStream_t stream) {
  const float* x  = (const float*)d_in[0];
  const float* Wq = (const float*)d_in[1];
  const float* bq = (const float*)d_in[2];
  const float* Wk = (const float*)d_in[3];
  const float* bk = (const float*)d_in[4];
  const float* Wv = (const float*)d_in[5];
  const float* bv = (const float*)d_in[6];
  const float* Wo = (const float*)d_in[7];
  const float* bo = (const float*)d_in[8];

  char* ws = (char*)d_ws;
  size_t off = 0;
  bf16_t* xb  = (bf16_t*)(ws + off); off += (size_t)BSROWS * EMBED * 2;  // 8 MB
  bf16_t* wqp = (bf16_t*)(ws + off); off += (size_t)EMBED * EMBED * 2;   // 2 MB
  bf16_t* wkp = (bf16_t*)(ws + off); off += (size_t)EMBED * EMBED * 2;
  bf16_t* wvp = (bf16_t*)(ws + off); off += (size_t)EMBED * EMBED * 2;
  bf16_t* wop = (bf16_t*)(ws + off); off += (size_t)EMBED * EMBED * 2;
  bf16_t* qb  = (bf16_t*)(ws + off); off += (size_t)BSROWS * EMBED * 2;
  bf16_t* kb_ = (bf16_t*)(ws + off); off += (size_t)BSROWS * EMBED * 2;
  bf16_t* vtg = (bf16_t*)(ws + off); off += (size_t)BSROWS * EMBED * 2;  // [b][h][d][s]
  bf16_t* ab  = (bf16_t*)(ws + off); off += (size_t)BSROWS * EMBED * 2;

  const int nx = BSROWS * EMBED;
  cvt_f32_to_bf16<<<nx / 256, 256, 0, stream>>>(x, xb, nx);
  const int nw = EMBED * EMBED;
  cvt_pack_weight<<<nw / 256, 256, 0, stream>>>(Wq, wqp, EMBED);
  cvt_pack_weight<<<nw / 256, 256, 0, stream>>>(Wk, wkp, EMBED);
  cvt_pack_weight<<<nw / 256, 256, 0, stream>>>(Wv, wvp, EMBED);
  cvt_pack_weight<<<nw / 256, 256, 0, stream>>>(Wo, wop, EMBED);

  dim3 gg(EMBED / 64, BSROWS / 64);   // (16, 64)
  gemm_bf16_wmma<1><<<gg, 256, 0, stream>>>(xb, wqp, bq, nullptr, qb,  BSROWS, EMBED, EMBED);
  gemm_bf16_wmma<1><<<gg, 256, 0, stream>>>(xb, wkp, bk, nullptr, kb_, BSROWS, EMBED, EMBED);
  gemm_bf16_wmma<2><<<gg, 256, 0, stream>>>(xb, wvp, bv, nullptr, vtg, BSROWS, EMBED, EMBED);

  alibi_flash_attn<<<BATCH * NHEADS * (SEQ / 128), 256, 0, stream>>>(qb, kb_, vtg, ab);

  gemm_bf16_wmma<0><<<gg, 256, 0, stream>>>(ab, wop, bo, (float*)d_out, nullptr,
                                            BSROWS, EMBED, EMBED);
}
